// PhysicsConsistencyLoss_26164940767824
// MI455X (gfx1250) — compile-verified, matrix operand order checked by reading, not run on hardware
//
#include <hip/hip_runtime.h>
#include <math.h>

typedef __attribute__((ext_vector_type(2)))  float    v2f;
typedef __attribute__((ext_vector_type(8)))  float    v8f;
typedef __attribute__((ext_vector_type(16))) _Float16 v16h;

#define PB 16
#define PT 100
#define PN 128

#if __has_builtin(__builtin_amdgcn_wmma_f32_16x16x4_f32)
#define USE_F32X4 1
#elif __has_builtin(__builtin_amdgcn_wmma_f32_16x16x32_f16)
#define USE_F16X32 1
#endif

__device__ __forceinline__ float fast_sqrtf(float x) {
#if __has_builtin(__builtin_amdgcn_sqrtf)
    return __builtin_amdgcn_sqrtf(x);
#else
    return sqrtf(x);
#endif
}

__device__ __forceinline__ void block_reduce_atomic(float partial, float* accum,
                                                    float* sred) {
    const int tid = threadIdx.x;
    sred[tid] = partial;
    __syncthreads();
    for (int s = 128; s > 0; s >>= 1) {
        if (tid < s) sred[tid] += sred[tid + s];
        __syncthreads();
    }
    if (tid == 0) atomicAdd(accum, sred[0]);
}

#if defined(USE_F32X4)
// One 16x16 tile of the Gram matrix via V_WMMA_F32_16X16X4_F32, plus the
// distance/relu epilogue. MASKED=true only for the diagonal tile (ti==tj);
// off-diagonal tiles are entirely i<j.
template <bool MASKED>
__device__ __forceinline__ float tile_pen(v2f a, int tj, int m, bool lo, int ibase,
                                          const float* __restrict__ spos,
                                          const float* __restrict__ srr,
                                          const float* __restrict__ ri) {
    const int rowB = tj * 16 + m;  // B[k][n] = P[tj*16+n][k]
    // spos[row*4+3] == 0, so address-select gives zeros for the K=3 pad lane
    // with unconditional LDS loads (no exec-masked load regions).
    v2f bm;
    bm.x = spos[rowB * 4 + (lo ? 0 : 2)];
    bm.y = spos[rowB * 4 + (lo ? 1 : 3)];

    v8f c = {};
    c = __builtin_amdgcn_wmma_f32_16x16x4_f32(false, a, false, bm,
                                              (short)0, c, false, false);

    const int j    = tj * 16 + m;  // global column index for this lane
    const float rj = srr[j];
    float p = 0.0f;
#pragma unroll
    for (int v = 0; v < 8; ++v) {
        float d2   = ri[v] + rj - 2.0f * c[v];
        float dist = fast_sqrtf(fmaxf(d2, 0.0f));
        float pen  = fmaxf(0.05f - dist, 0.0f);
        if (MASKED) pen = (ibase + v < j) ? pen : 0.0f;  // branchless cndmask
        p += pen;
    }
    return p;
}
#endif

// ---------------------------------------------------------------------------
// Penetration loss: one block per (b,t) frame. Gram matrix G = P * P^T via
// WMMA (K=3 padded to 4), d2 = r_i + r_j - 2*G_ij, pen = relu(0.05 - dist),
// upper-triangle only (36 of 64 tiles per frame).
// ---------------------------------------------------------------------------
__global__ __launch_bounds__(256) void pen_kernel(const float* __restrict__ traj,
                                                  float* __restrict__ accum) {
    __shared__ float spos[PN * 4];  // xyz0 padded (pad slot guaranteed zero)
    __shared__ float srr[PN];       // |p|^2
    __shared__ float sred[256];

    const int tid   = threadIdx.x;
    const long base = (long)blockIdx.x * (PN * 3);

    __builtin_prefetch(traj + base, 0, 1);  // global_prefetch_b8

    if (tid < PN) {
        float x = traj[base + tid * 3 + 0];
        float y = traj[base + tid * 3 + 1];
        float z = traj[base + tid * 3 + 2];
        spos[tid * 4 + 0] = x;
        spos[tid * 4 + 1] = y;
        spos[tid * 4 + 2] = z;
        spos[tid * 4 + 3] = 0.0f;
        srr[tid] = x * x + y * y + z * z;
    }
    __syncthreads();

    const int lane = tid & 31;
    const int ti   = __builtin_amdgcn_readfirstlane(tid >> 5);  // wave id = tile row
    const int m    = lane & 15;
    const bool lo  = lane < 16;

    float partial = 0.0f;

#if defined(USE_F32X4)
    // A-matrix 16x4 f32: lanes 0-15 hold K=0(V0),K=1(V1); lanes 16-31 K=2,K=3.
    const int rowA = ti * 16 + m;
    v2f a;
    a.x = spos[rowA * 4 + (lo ? 0 : 2)];
    a.y = spos[rowA * 4 + (lo ? 1 : 3)];  // pad slot -> 0 for hi half

    // Row-norm terms for this lane's 8 output rows; invariant across tj.
    const int ibase = ti * 16 + (lo ? 0 : 8);
    float ri[8];
#pragma unroll
    for (int v = 0; v < 8; ++v) ri[v] = srr[ibase + v];

    partial += tile_pen<true>(a, ti, m, lo, ibase, spos, srr, ri);  // diagonal
    for (int tj = ti + 1; tj < 8; ++tj)
        partial += tile_pen<false>(a, tj, m, lo, ibase, spos, srr, ri);
#elif defined(USE_F16X32)
    // f16 fallback (not expected on gfx1250): K padded 3 -> 32.
    const int rowA = ti * 16 + m;
    v16h a = {};
    if (lo) {
        a[0] = (_Float16)spos[rowA * 4 + 0];
        a[1] = (_Float16)spos[rowA * 4 + 1];
        a[2] = (_Float16)spos[rowA * 4 + 2];
    }
    const int ibase = ti * 16 + (lo ? 0 : 8);
    for (int tj = ti; tj < 8; ++tj) {
        const int rowB = tj * 16 + m;
        v16h bm = {};
        if (lo) {
            bm[0] = (_Float16)spos[rowB * 4 + 0];
            bm[1] = (_Float16)spos[rowB * 4 + 1];
            bm[2] = (_Float16)spos[rowB * 4 + 2];
        }
        v8f c = {};
        c = __builtin_amdgcn_wmma_f32_16x16x32_f16(false, a, false, bm,
                                                   (short)0, c, false, false);
        const int j    = tj * 16 + m;
        const float rj = srr[j];
#pragma unroll
        for (int v = 0; v < 8; ++v) {
            float d2   = srr[ibase + v] + rj - 2.0f * c[v];
            float dist = fast_sqrtf(fmaxf(d2, 0.0f));
            float pen  = fmaxf(0.05f - dist, 0.0f);
            partial += (ibase + v < j) ? pen : 0.0f;
        }
    }
#else
    // Host-parse / safety scalar path (never the device path on gfx1250).
    const int ibase = ti * 16 + (lo ? 0 : 8);
    for (int tj = ti; tj < 8; ++tj) {
        const int j    = tj * 16 + m;
        const float rj = srr[j];
        for (int v = 0; v < 8; ++v) {
            const int i = ibase + v;
            float g = spos[i * 4 + 0] * spos[j * 4 + 0] +
                      spos[i * 4 + 1] * spos[j * 4 + 1] +
                      spos[i * 4 + 2] * spos[j * 4 + 2];
            float d2   = srr[i] + rj - 2.0f * g;
            float dist = fast_sqrtf(fmaxf(d2, 0.0f));
            float pen  = fmaxf(0.05f - dist, 0.0f);
            partial += (i < j) ? pen : 0.0f;
        }
    }
#endif

    block_reduce_atomic(partial, accum, sred);
}

// ---------------------------------------------------------------------------
// Stability: sum of ||v|| over b, t in [95,100), n.  10240 elements.
// ---------------------------------------------------------------------------
__global__ __launch_bounds__(256) void stab_kernel(const float* __restrict__ vel,
                                                   float* __restrict__ accum) {
    __shared__ float sred[256];
    const int idx = blockIdx.x * 256 + threadIdx.x;
    float val = 0.0f;
    if (idx < PB * 5 * PN) {
        const int n    = idx % PN;
        const int rest = idx / PN;
        const int t    = 95 + (rest % 5);
        const int b    = rest / 5;
        const long off = (((long)b * PT + t) * PN + n) * 3;
        float x = vel[off + 0], y = vel[off + 1], z = vel[off + 2];
        val = fast_sqrtf(x * x + y * y + z * z);
    }
    block_reduce_atomic(val, accum, sred);
}

// ---------------------------------------------------------------------------
// Kinetic energy difference: sum over (b,n) of (|v_end|^2 - |v_start|^2).
// ---------------------------------------------------------------------------
__global__ __launch_bounds__(256) void ke_kernel(const float* __restrict__ vel,
                                                 float* __restrict__ accum) {
    __shared__ float sred[256];
    const int idx = blockIdx.x * 256 + threadIdx.x;
    float val = 0.0f;
    if (idx < PB * PN) {
        const int n = idx % PN;
        const int b = idx / PN;
        const long o0 = (((long)b * PT + 0) * PN + n) * 3;
        const long o1 = (((long)b * PT + (PT - 1)) * PN + n) * 3;
        float x0 = vel[o0], y0 = vel[o0 + 1], z0 = vel[o0 + 2];
        float x1 = vel[o1], y1 = vel[o1 + 1], z1 = vel[o1 + 2];
        val = (x1 * x1 + y1 * y1 + z1 * z1) - (x0 * x0 + y0 * y0 + z0 * z0);
    }
    block_reduce_atomic(val, accum, sred);
}

// ---------------------------------------------------------------------------
// Work: sum over (b, t<T-1, n) of F[b,t,n] . (P[b,t+1,n] - P[b,t,n]).
// ---------------------------------------------------------------------------
__global__ __launch_bounds__(256) void work_kernel(const float* __restrict__ traj,
                                                   const float* __restrict__ frc,
                                                   float* __restrict__ accum) {
    __shared__ float sred[256];
    const int idx = blockIdx.x * 256 + threadIdx.x;
    float val = 0.0f;
    if (idx < PB * (PT - 1) * PN) {
        const int n    = idx % PN;
        const int rest = idx / PN;
        const int t    = rest % (PT - 1);
        const int b    = rest / (PT - 1);
        const long o0 = (((long)b * PT + t) * PN + n) * 3;
        const long o1 = o0 + (long)PN * 3;  // t+1
        float dx = traj[o1 + 0] - traj[o0 + 0];
        float dy = traj[o1 + 1] - traj[o0 + 1];
        float dz = traj[o1 + 2] - traj[o0 + 2];
        val = frc[o0 + 0] * dx + frc[o0 + 1] * dy + frc[o0 + 2] * dz;
    }
    block_reduce_atomic(val, accum, sred);
}

__global__ void zero_kernel(float* ws) {
    if (threadIdx.x < 8) ws[threadIdx.x] = 0.0f;
}

__global__ void finalize_kernel(const float* __restrict__ ws,
                                float* __restrict__ out) {
    if (threadIdx.x == 0) {
        const float pen  = ws[0] / (16.0f * 100.0f * 8128.0f); // /(B*T*N*(N-1)/2)
        const float stab = ws[1] / (16.0f * 5.0f * 128.0f);
        const float ke   = 0.5f * ws[2] / (16.0f * 128.0f);
        const float work = ws[3] / (16.0f * 99.0f * 128.0f);
        out[0] = 10.0f * pen + 1.0f * stab + 0.1f * fabsf(ke - work);
    }
}

extern "C" void kernel_launch(void* const* d_in, const int* in_sizes, int n_in,
                              void* d_out, int out_size, void* d_ws, size_t ws_size,
                              hipStream_t stream) {
    const float* traj = (const float*)d_in[0];
    const float* vel  = (const float*)d_in[1];
    const float* frc  = (const float*)d_in[2];
    float* out = (float*)d_out;
    float* ws  = (float*)d_ws;  // [0]=pen, [1]=stab, [2]=ke diff, [3]=work

    zero_kernel<<<1, 8, 0, stream>>>(ws);

    pen_kernel<<<PB * PT, 256, 0, stream>>>(traj, ws + 0);

    stab_kernel<<<(PB * 5 * PN + 255) / 256, 256, 0, stream>>>(vel, ws + 1);

    ke_kernel<<<(PB * PN + 255) / 256, 256, 0, stream>>>(vel, ws + 2);

    work_kernel<<<(PB * (PT - 1) * PN + 255) / 256, 256, 0, stream>>>(traj, frc, ws + 3);

    finalize_kernel<<<1, 32, 0, stream>>>(ws, out);
}
